// MambaVectorField_57853209477734
// MI455X (gfx1250) — matrix operand order repaired and checked by reading, not compiled, not run on hardware
//
#include <hip/hip_runtime.h>
#include <hip/hip_bf16.h>

// ---------------------------------------------------------------------------
// MambaVectorField for gfx1250 (CDNA5):
//   bf16 WMMA GEMMs with async global->LDS double buffering + fused scan
// B=4, S=4096, D_MODEL=1024, D_INNER=2048, D_STATE=16
// ---------------------------------------------------------------------------

#define B_DIM      4
#define S_DIM      4096
#define D_MODEL    1024
#define D_INNER    2048
#define D_STATE    16
#define M_ROWS     (B_DIM * S_DIM)        // 16384

typedef __attribute__((ext_vector_type(16))) __bf16         bf16x16;
typedef __attribute__((ext_vector_type(8)))  float          f32x8;
typedef __attribute__((ext_vector_type(8)))  unsigned short u16x8;

union FragAB { bf16x16 v; u16x8 h[2]; };

__device__ __forceinline__ unsigned short f2bf(float f) {
  unsigned u = __float_as_uint(f);
  u += 0x7FFFu + ((u >> 16) & 1u);        // round-to-nearest-even
  return (unsigned short)(u >> 16);
}

__device__ __forceinline__ u16x8 pack8(float4 a, float4 b) {
  u16x8 r;
  r[0] = f2bf(a.x); r[1] = f2bf(a.y); r[2] = f2bf(a.z); r[3] = f2bf(a.w);
  r[4] = f2bf(b.x); r[5] = f2bf(b.y); r[6] = f2bf(b.z); r[7] = f2bf(b.w);
  return r;
}

__device__ __forceinline__ float silu(float x) { return x / (1.0f + __expf(-x)); }

// Async 16-byte global -> LDS copy (CDNA5 GLOBAL_LOAD_ASYNC_TO_LDS_B128,
// tracked by ASYNCcnt).  The low 32 bits of a generic pointer to a __shared__
// object are the LDS offset (flat->local is a truncate on amdgcn).
__device__ __forceinline__ void async_copy16(void* lds, const void* gptr) {
  unsigned           l = (unsigned)(uintptr_t)lds;
  unsigned long long g = (unsigned long long)(uintptr_t)gptr;
  asm volatile("global_load_async_to_lds_b128 %0, %1, off"
               :: "v"(l), "v"(g) : "memory");
}

// ---------------------------------------------------------------------------
// f32 -> bf16 conversion pass (8 elements / thread)
// ---------------------------------------------------------------------------
__global__ __launch_bounds__(256) void cvt_bf16_kernel(
    const float* __restrict__ s, unsigned short* __restrict__ d, int n8) {
  const int i = blockIdx.x * 256 + threadIdx.x;
  if (i < n8) {
    const float4 a = ((const float4*)s)[2 * i + 0];
    const float4 b = ((const float4*)s)[2 * i + 1];
    ((u16x8*)d)[i] = pack8(a, b);
  }
}

// ---------------------------------------------------------------------------
// WMMA GEMM:  C[M,N] = A[M,K] @ Bw[N,K]^T   (A, Bw pre-converted bf16)
// Block tile 128x128, K-step 32, 8 wave32s (wave tile 32x64), ping-pong LDS
// staged with async global->LDS copies (4 per wave per tile).
// EPI==0 : plain f32 store to out0
// EPI==1 : in_proj epilogue: col<2048 -> xr = acc + t_emb[b][col]
//                            col>=2048 -> gate = silu(acc)
// ---------------------------------------------------------------------------
template <int EPI>
__global__ __launch_bounds__(256) void wmma_gemm_kernel(
    const unsigned short* __restrict__ A, const unsigned short* __restrict__ Bw,
    float* __restrict__ out0, float* __restrict__ out1,
    const float* __restrict__ temb, int M, int N, int K) {
  __shared__ __align__(16) unsigned short As[2][128][40];  // bf16, padded rows
  __shared__ __align__(16) unsigned short Bs[2][128][40];

  const int tid = threadIdx.x;
  const int bm  = blockIdx.y * 128;
  const int bn  = blockIdx.x * 128;

  // staging role: 2 threads per tile row, 16 bf16 (2 x b128) each per matrix
  const int ldr = tid >> 1;                 // 0..127
  const int ldc = (tid & 1) << 4;           // ushort offset 0 / 16

  // wave/lane decomposition for WMMA
  const int lane = tid & 31;
  const int wm   = ((tid >> 6) & 3) * 32;   // wave>>1 -> m offset
  const int wn   = ((tid >> 5) & 1) * 64;   // wave&1  -> n offset
  const int l16  = lane & 15;
  const int hi   = lane >> 4;               // 0: lanes 0-15, 1: lanes 16-31
  const int klo  = hi << 3;                 // A frag K base 0 / 8
  const int kbh  = hi << 4;                 // unused; B frags use klo (0 / 8) like A

  f32x8 acc[2][4];
#pragma unroll
  for (int i = 0; i < 2; ++i)
#pragma unroll
    for (int j = 0; j < 4; ++j) acc[i][j] = (f32x8){};

  const unsigned short* pAr = A + (size_t)(bm + ldr) * K + ldc;
  const unsigned short* pBr = Bw + (size_t)(bn + ldr) * K + ldc;

  // prologue: stage tile 0 into buffer 0
  async_copy16(&As[0][ldr][ldc + 0], pAr + 0);
  async_copy16(&As[0][ldr][ldc + 8], pAr + 8);
  async_copy16(&Bs[0][ldr][ldc + 0], pBr + 0);
  async_copy16(&Bs[0][ldr][ldc + 8], pBr + 8);

  for (int k0 = 0; k0 < K; k0 += 32) {
    const int cur = (k0 >> 5) & 1;
    if (k0 + 32 < K) {  // stage next tile into the other buffer, then wait
      const int nxt = cur ^ 1;
      async_copy16(&As[nxt][ldr][ldc + 0], pAr + k0 + 32 + 0);
      async_copy16(&As[nxt][ldr][ldc + 8], pAr + k0 + 32 + 8);
      async_copy16(&Bs[nxt][ldr][ldc + 0], pBr + k0 + 32 + 0);
      async_copy16(&Bs[nxt][ldr][ldc + 8], pBr + k0 + 32 + 8);
      asm volatile("s_wait_asynccnt 0x4" ::: "memory");  // cur tile landed
    } else {
      asm volatile("s_wait_asynccnt 0x0" ::: "memory");
    }
    __syncthreads();

    // ---- LDS -> fragments per CDNA5 WMMA lane layout ----
    FragAB fa[2], fb[4];
#pragma unroll
    for (int tm = 0; tm < 2; ++tm) {
      const int r = wm + tm * 16 + l16;     // A row (M)
      fa[tm].h[0] = *(const u16x8*)&As[cur][r][klo];       // K = klo..klo+7
      fa[tm].h[1] = *(const u16x8*)&As[cur][r][16 + klo];  // K = 16+klo..
    }
#pragma unroll
    for (int tn = 0; tn < 4; ++tn) {
      const int r = wn + tn * 16 + l16;     // B col (N) stored as row of Bs
      fb[tn].h[0] = *(const u16x8*)&Bs[cur][r][klo];       // K = klo..klo+7
      fb[tn].h[1] = *(const u16x8*)&Bs[cur][r][16 + klo];  // K = 16+klo..
    }

#pragma unroll
    for (int tm = 0; tm < 2; ++tm)
#pragma unroll
      for (int tn = 0; tn < 4; ++tn)
        acc[tm][tn] = __builtin_amdgcn_wmma_f32_16x16x32_bf16(
            false, fa[tm].v, false, fb[tn].v, (short)0, acc[tm][tn], false,
            false);
    __syncthreads();  // all waves done reading `cur` before it is re-staged
  }

  // ---- epilogue (C layout: VGPR r -> M = r + 8*hi, N = l16) ----
#pragma unroll
  for (int tm = 0; tm < 2; ++tm) {
#pragma unroll
    for (int tn = 0; tn < 4; ++tn) {
#pragma unroll
      for (int r = 0; r < 8; ++r) {
        const int gm = bm + wm + tm * 16 + r + hi * 8;
        const int gn = bn + wn + tn * 16 + l16;
        float v = acc[tm][tn][r];
        if (EPI == 0) {
          out0[(size_t)gm * N + gn] = v;
        } else {
          if (gn < D_INNER) {
            const int b = gm >> 12;  // row / S
            out0[(size_t)gm * D_INNER + gn] = v + temb[b * D_INNER + gn];
          } else {
            out1[(size_t)gm * D_INNER + (gn - D_INNER)] = silu(v);
          }
        }
      }
    }
  }
}

// ---------------------------------------------------------------------------
// Causal depthwise conv1d(k=4) + SiLU + LayerNorm + clamp(||xn||,1)
// ---------------------------------------------------------------------------
__global__ __launch_bounds__(256) void conv_silu_ln_kernel(
    const float* __restrict__ xr, const float* __restrict__ Wc,
    const float* __restrict__ bc, const float* __restrict__ lng,
    const float* __restrict__ lnb, float* __restrict__ xn,
    float* __restrict__ normc) {
  const int row = blockIdx.x;               // b*S + s
  const int b   = row >> 12;
  const int s   = row & (S_DIM - 1);
  const int tid = threadIdx.x;

  __shared__ float2 red[256];
  float xv[8];
  float lsum = 0.f, lsq = 0.f;
#pragma unroll
  for (int i = 0; i < 8; ++i) {
    const int ch = tid + i * 256;
    float a = bc[ch];
#pragma unroll
    for (int k = 0; k < 4; ++k) {
      const int sp = s - 3 + k;
      if (sp >= 0)
        a += Wc[ch * 4 + k] * xr[((size_t)b * S_DIM + sp) * D_INNER + ch];
    }
    a = silu(a);
    xv[i] = a;
    lsum += a;
    lsq += a * a;
  }
  red[tid] = make_float2(lsum, lsq);
  __syncthreads();
  for (int off = 128; off > 0; off >>= 1) {
    if (tid < off) {
      red[tid].x += red[tid + off].x;
      red[tid].y += red[tid + off].y;
    }
    __syncthreads();
  }
  const float mu   = red[0].x * (1.0f / D_INNER);
  const float var  = red[0].y * (1.0f / D_INNER) - mu * mu;
  const float rstd = rsqrtf(var + 1e-5f);
  __syncthreads();

  float l2 = 0.f;
#pragma unroll
  for (int i = 0; i < 8; ++i) {
    const int ch = tid + i * 256;
    float v = (xv[i] - mu) * rstd * lng[ch] + lnb[ch];
    xn[(size_t)row * D_INNER + ch] = v;
    l2 += v * v;
  }
  red[tid].x = l2;
  __syncthreads();
  for (int off = 128; off > 0; off >>= 1) {
    if (tid < off) red[tid].x += red[tid + off].x;
    __syncthreads();
  }
  if (tid == 0) normc[row] = fminf(sqrtf(red[0].x), 1.0f);
}

// ---------------------------------------------------------------------------
// xn @ W_xproj^T (48 outputs) + dt-transform -> Ad, Bd, Cm  (per row)
// ---------------------------------------------------------------------------
__global__ __launch_bounds__(64) void xproj_kernel(
    const float* __restrict__ xn, const float* __restrict__ Wx,
    const float* __restrict__ Wdt, const float* __restrict__ bdt,
    const float* __restrict__ Alog, const float* __restrict__ normc,
    float* __restrict__ Ad, float* __restrict__ Bd, float* __restrict__ Cm) {
  const int row = blockIdx.x;
  const int tid = threadIdx.x;
  __shared__ float proj[48];

  if (tid < 48) {
    const float* xp = xn + (size_t)row * D_INNER;
    const float* wp = Wx + (size_t)tid * D_INNER;
    float acc = 0.f;
    for (int k = 0; k < D_INNER; k += 4) {
      float4 a = *(const float4*)(xp + k);
      float4 w = *(const float4*)(wp + k);
      acc += a.x * w.x + a.y * w.y + a.z * w.z + a.w * w.w;
    }
    proj[tid] = acc;
  }
  __syncthreads();
  if (tid < D_STATE) {
    float acc = bdt[tid];
#pragma unroll
    for (int i = 0; i < D_STATE; ++i) acc += Wdt[tid * D_STATE + i] * proj[32 + i];
    const float dt = tanhf(acc) * 0.01f;
    const float Aw = -tanhf(Alog[tid]);
    Ad[(size_t)row * D_STATE + tid] = 1.0f + dt * Aw;
    Bd[(size_t)row * D_STATE + tid] = proj[tid] * normc[row];
    Cm[(size_t)row * D_STATE + tid] = proj[16 + tid];
  }
}

// ---------------------------------------------------------------------------
// Selective scan fused with  z = (y + D*xn) * silu_gate, output stored bf16
// so the out_proj GEMM consumes it directly.  One wave32 per 32 channels;
// Ad/Bd/Cm rows are wave-uniform -> scalar loads; xn/gate coalesced.
// ---------------------------------------------------------------------------
__global__ __launch_bounds__(32) void scan_kernel(
    const float* __restrict__ xn, const float* __restrict__ gate,
    const float* __restrict__ Ad, const float* __restrict__ Bd,
    const float* __restrict__ Cm, const float* __restrict__ Dv,
    unsigned short* __restrict__ zb) {
  const int b  = blockIdx.y;
  const int ch = blockIdx.x * 32 + threadIdx.x;
  const size_t rowbase = (size_t)b * S_DIM;
  const float dch = Dv[ch];

  float h[D_STATE];
#pragma unroll
  for (int i = 0; i < D_STATE; ++i) h[i] = 0.f;

  for (int s = 0; s < S_DIM; ++s) {
    const size_t row = rowbase + s;
    const float* ap = Ad + row * D_STATE;   // uniform across wave
    const float* bp = Bd + row * D_STATE;
    const float* cp = Cm + row * D_STATE;
    const float xt = xn[row * D_INNER + ch];
    const float gt = gate[row * D_INNER + ch];

    if (s + 8 < S_DIM) {
      __builtin_prefetch(xn + (row + 8) * D_INNER + ch, 0, 1);
      __builtin_prefetch(gate + (row + 8) * D_INNER + ch, 0, 1);
    }

    float y = 0.f;
#pragma unroll
    for (int i = 0; i < D_STATE; ++i) {
      float hv = h[i] * ap[i] + bp[i] * xt;
      hv = fminf(fmaxf(hv, -10.0f), 10.0f);
      h[i] = hv;
      y += cp[i] * hv;
    }
    zb[row * D_INNER + ch] = f2bf((y + dch * xt) * gt);
  }
}

// ---------------------------------------------------------------------------
extern "C" void kernel_launch(void* const* d_in, const int* in_sizes, int n_in,
                              void* d_out, int out_size, void* d_ws,
                              size_t ws_size, hipStream_t stream) {
  const float* x      = (const float*)d_in[0];   // [4,4096,1024]
  const float* t_emb  = (const float*)d_in[1];   // [4,2048]
  const float* W_in   = (const float*)d_in[2];   // [4096,1024]
  const float* W_conv = (const float*)d_in[3];   // [2048,1,4]
  const float* b_conv = (const float*)d_in[4];   // [2048]
  const float* ln_g   = (const float*)d_in[5];   // [2048]
  const float* ln_b   = (const float*)d_in[6];   // [2048]
  const float* W_xp   = (const float*)d_in[7];   // [48,2048]
  const float* W_dt   = (const float*)d_in[8];   // [16,16]
  const float* b_dt   = (const float*)d_in[9];   // [16]
  const float* A_log  = (const float*)d_in[10];  // [16]
  const float* Dvec   = (const float*)d_in[11];  // [2048]
  const float* W_out  = (const float*)d_in[12];  // [1024,2048]
  float* out = (float*)d_out;                    // [4,4096,1024]

  char* ws = (char*)d_ws;
  const size_t BIG = (size_t)M_ROWS * D_INNER * sizeof(float);  // 128 MiB
  float* xr    = (float*)(ws);                  // f32 stream; dead after conv
  float* gate  = (float*)(ws + BIG);            // silu(gate), f32
  float* xn    = (float*)(ws + 2 * BIG);        // normalized stream, f32
  float* normc = (float*)(ws + 3 * BIG);        // [M]
  size_t o = 3 * BIG + (size_t)M_ROWS * 4;
  float* Ad = (float*)(ws + o); o += (size_t)M_ROWS * D_STATE * 4;
  float* Bd = (float*)(ws + o); o += (size_t)M_ROWS * D_STATE * 4;
  float* Cm = (float*)(ws + o); o += (size_t)M_ROWS * D_STATE * 4;
  unsigned short* xbf  = (unsigned short*)(ws + o); o += (size_t)M_ROWS * D_MODEL * 2;
  unsigned short* Wibf = (unsigned short*)(ws + o); o += (size_t)2 * D_INNER * D_MODEL * 2;
  unsigned short* Wobf = (unsigned short*)(ws + o); o += (size_t)D_MODEL * D_INNER * 2;
  unsigned short* zbf  = (unsigned short*)(ws);  // overlays dead xr region

  // 0) one-shot f32 -> bf16 conversions feeding the WMMA GEMMs
  {
    const int nx = M_ROWS * D_MODEL / 8;              // 2097152
    cvt_bf16_kernel<<<(nx + 255) / 256, 256, 0, stream>>>(x, xbf, nx);
    const int nwi = 2 * D_INNER * D_MODEL / 8;        // 524288
    cvt_bf16_kernel<<<(nwi + 255) / 256, 256, 0, stream>>>(W_in, Wibf, nwi);
    const int nwo = D_MODEL * D_INNER / 8;            // 262144
    cvt_bf16_kernel<<<(nwo + 255) / 256, 256, 0, stream>>>(W_out, Wobf, nwo);
  }
  // 1) in_proj GEMM: [16384,1024] @ [4096,1024]^T, fused +t_emb / silu(gate)
  {
    dim3 grid((2 * D_INNER) / 128, M_ROWS / 128);
    wmma_gemm_kernel<1><<<grid, 256, 0, stream>>>(
        xbf, Wibf, xr, gate, t_emb, M_ROWS, 2 * D_INNER, D_MODEL);
  }
  // 2) causal conv + SiLU + LayerNorm + ||xn|| clamp
  conv_silu_ln_kernel<<<M_ROWS, 256, 0, stream>>>(xr, W_conv, b_conv, ln_g,
                                                  ln_b, xn, normc);
  // 3) xproj + dt transform -> Ad, Bd, Cm
  xproj_kernel<<<M_ROWS, 64, 0, stream>>>(xn, W_xp, W_dt, b_dt, A_log, normc,
                                          Ad, Bd, Cm);
  // 4) selective scan fused with output gating -> zbf (bf16)
  {
    dim3 grid(D_INNER / 32, B_DIM);
    scan_kernel<<<grid, 32, 0, stream>>>(xn, gate, Ad, Bd, Cm, Dvec, zbf);
  }
  // 5) out_proj GEMM: [16384,2048] @ [1024,2048]^T -> d_out
  {
    dim3 grid(D_MODEL / 128, M_ROWS / 128);
    wmma_gemm_kernel<0><<<grid, 256, 0, stream>>>(
        zbf, Wobf, out, nullptr, nullptr, M_ROWS, D_MODEL, D_INNER);
  }
}
